// VectorQuantizer_30142080484148
// MI455X (gfx1250) — matrix operand order repaired and checked by reading, not compiled, not run on hardware
//
#include <hip/hip_runtime.h>
#include <hip/hip_bf16.h>

// Problem constants (match reference)
#define BB 8
#define NN 1024
#define CC 256
#define KK 4096
#define SN 11
#define NUMEL (BB * NN * CC)   // 2,097,152

typedef __attribute__((ext_vector_type(16))) __bf16          v16bf;
typedef __attribute__((ext_vector_type(16))) unsigned short  v16u;
typedef __attribute__((ext_vector_type(8)))  float           v8f;

// ---------- bf16 helpers (RNE, bit-level) ----------
__device__ inline unsigned short f2bf(float x) {
    union { float f; unsigned u; } v; v.f = x;
    unsigned u = v.u;
    u += 0x7FFFu + ((u >> 16) & 1u);
    return (unsigned short)(u >> 16);
}
__device__ inline float bf2f(unsigned short h) {
    union { unsigned u; float f; } v; v.u = ((unsigned)h) << 16;
    return v.f;
}

// ---------------- init: f_hat(d_out)=0, f_rest=f, loss acc=0 ----------------
__global__ void k_init(float* __restrict__ out, const float* __restrict__ f,
                       float* __restrict__ f_rest, float* __restrict__ qacc) {
    int i = blockIdx.x * 256 + threadIdx.x;
    out[i] = 0.0f;
    f_rest[i] = f[i];
    if (i == 0) *qacc = 0.0f;
}

// ---------------- embedding = base @ proj_w^T + proj_b ; e_sq ; bf16-split transpose ----
__global__ void k_emb(const float* __restrict__ base, const float* __restrict__ pw,
                      const float* __restrict__ pb, float* __restrict__ emb,
                      unsigned short* __restrict__ embT_hi, unsigned short* __restrict__ embT_lo,
                      float* __restrict__ esq) {
    __shared__ float bs[CC];
    __shared__ float red[256];
    int k = blockIdx.x;          // codebook row
    int c = threadIdx.x;         // channel
    bs[c] = base[k * CC + c];
    __syncthreads();
    float acc = pb[c];
    const float* wrow = pw + c * CC;
    #pragma unroll 8
    for (int j = 0; j < CC; ++j) acc = fmaf(bs[j], wrow[j], acc);
    emb[k * CC + c] = acc;
    unsigned short hi = f2bf(acc);
    float lo = acc - bf2f(hi);
    embT_hi[c * KK + k] = hi;
    embT_lo[c * KK + k] = f2bf(lo);
    // e_sq block reduction
    red[c] = acc * acc;
    __syncthreads();
    for (int s = 128; s > 0; s >>= 1) {
        if (c < s) red[c] += red[c + s];
        __syncthreads();
    }
    if (c == 0) esq[k] = red[0];
}

// ---------------- pool: q[m][c] = mean over segment of f_rest (bf16 hi/lo split) -------
__global__ void k_pool(const float* __restrict__ f_rest,
                       unsigned short* __restrict__ q_hi, unsigned short* __restrict__ q_lo,
                       int pn, int M) {
    int m = blockIdx.x;                 // padded row id
    int msrc = m < M ? m : M - 1;       // clamp pad rows
    int c = threadIdx.x;
    int b = msrc / pn, p = msrc % pn;
    int seg = NN / pn;
    const float* src = f_rest + ((size_t)b * NN + (size_t)p * seg) * CC + c;
    float s = 0.0f;
    for (int t = 0; t < seg; ++t) s += src[(size_t)t * CC];
    s *= (1.0f / (float)seg);
    unsigned short hi = f2bf(s);
    q_hi[(size_t)m * CC + c] = hi;
    q_lo[(size_t)m * CC + c] = f2bf(s - bf2f(hi));
}

// ---------------- dist+argmin: WMMA bf16 split GEMM, argmin over K=4096 ----------------
// block = 256 threads = 8 waves. Each block owns a 16-row tile; waves split the 256
// column tiles of 16 codes each. launch_bounds(256,1) keeps the 128 resident
// A-fragment VGPRs (hi+lo) out of scratch.
__global__ void __launch_bounds__(256, 1)
k_dist(const unsigned short* __restrict__ q_hi,
       const unsigned short* __restrict__ q_lo,
       const unsigned short* __restrict__ embT_hi,
       const unsigned short* __restrict__ embT_lo,
       const float* __restrict__ esq,
       int* __restrict__ idxOut, int M) {
    __shared__ float cV[8][16];
    __shared__ int   cI[8][16];

    const int tid  = threadIdx.x;
    const int wave = tid >> 5;
    const int lane = tid & 31;
    const int half = lane >> 4;       // K-half selector
    const int l    = lane & 15;       // row (A) / column-slot (D)
    const int mr   = blockIdx.x * 16;

    // ---- load A fragments (16 rows x 256 contraction, bf16 hi & lo), resident in VGPRs
    // ISA 16-bit A 16x32 layout: lane l holds row l; VGPR v holds K = kc*32 + half*8 + (v&3)*2 + (v>>2)*16
    const int arow = (mr + l) < M ? (mr + l) : (M - 1);
    const unsigned short* qh = q_hi + (size_t)arow * CC;
    const unsigned short* ql = q_lo + (size_t)arow * CC;
    v16bf aH[8], aL[8];
    #pragma unroll
    for (int kc = 0; kc < 8; ++kc) {
        v16u tH, tL;
        #pragma unroll
        for (int v = 0; v < 8; ++v) {
            int k0 = kc * 32 + half * 8 + (v & 3) * 2 + (v >> 2) * 16;
            tH[2 * v]     = qh[k0];
            tH[2 * v + 1] = qh[k0 + 1];
            tL[2 * v]     = ql[k0];
            tL[2 * v + 1] = ql[k0 + 1];
        }
        aH[kc] = __builtin_bit_cast(v16bf, tH);
        aL[kc] = __builtin_bit_cast(v16bf, tL);
    }

    float bestV[8];
    int   bestI[8];
    #pragma unroll
    for (int r = 0; r < 8; ++r) { bestV[r] = 3.402823466e+38f; bestI[r] = 0; }

    // ---- iterate column tiles of 16 codes; wave w takes tiles w, w+8, ...
    for (int kt = wave * 16; kt < KK; kt += 8 * 16) {
        v8f acc = {};
        #pragma unroll
        for (int kc = 0; kc < 8; ++kc) {
            // B 32x16 layout: lane holds column kt+l; element s holds K = kc*32 + half*8 + (s&7) + (s>>3)*16
            int crow = kc * 32;
            const unsigned short* bhp = embT_hi + (size_t)crow * KK + kt + l;
            const unsigned short* blp = embT_lo + (size_t)crow * KK + kt + l;
            // Speculative prefetch of the next kt tile (unconditional: dropped if
            // past the region; +256B still lands inside our workspace). No exec-mask
            // branch in the hot loop.
            __builtin_prefetch(bhp + 128, 0, 3);
            __builtin_prefetch(blp + 128, 0, 3);
            v16u tbh, tbl;
            #pragma unroll
            for (int s = 0; s < 16; ++s) {
                size_t ko = (size_t)(half * 8 + (s & 7) + (s >> 3) * 16) * KK;
                tbh[s] = bhp[ko];
                tbl[s] = blp[ko];
            }
            v16bf bh = __builtin_bit_cast(v16bf, tbh);
            acc = __builtin_amdgcn_wmma_f32_16x16x32_bf16(false, aH[kc], false, bh,
                                                          (short)0, acc, false, false);
            acc = __builtin_amdgcn_wmma_f32_16x16x32_bf16(false, aL[kc], false, bh,
                                                          (short)0, acc, false, false);
            v16bf bl = __builtin_bit_cast(v16bf, tbl);
            acc = __builtin_amdgcn_wmma_f32_16x16x32_bf16(false, aH[kc], false, bl,
                                                          (short)0, acc, false, false);
        }
        // D layout: VGPR r -> row (r + 8*half), column (kt + l). ||q||^2 dropped (row-constant).
        float ek = esq[kt + l];
        #pragma unroll
        for (int r = 0; r < 8; ++r) {
            float d = fmaf(-2.0f, acc[r], ek);
            int   ki = kt + l;
            if (d < bestV[r]) { bestV[r] = d; bestI[r] = ki; }
        }
    }

    // ---- reduce across the 16 column slots (stays within each 16-lane half)
    #pragma unroll
    for (int r = 0; r < 8; ++r) {
        #pragma unroll
        for (int off = 1; off < 16; off <<= 1) {
            float ov = __shfl_xor(bestV[r], off, 32);
            int   oi = __shfl_xor(bestI[r], off, 32);
            if (ov < bestV[r] || (ov == bestV[r] && oi < bestI[r])) {
                bestV[r] = ov; bestI[r] = oi;
            }
        }
    }
    if (l == 0) {
        #pragma unroll
        for (int r = 0; r < 8; ++r) {
            cV[wave][r + 8 * half] = bestV[r];
            cI[wave][r + 8 * half] = bestI[r];
        }
    }
    __syncthreads();
    // ---- cross-wave reduce: 16 threads, one per row
    if (tid < 16) {
        float bv = cV[0][tid];
        int   bi = cI[0][tid];
        #pragma unroll
        for (int w = 1; w < 8; ++w) {
            float v = cV[w][tid];
            int   i = cI[w][tid];
            if (v < bv || (v == bv && i < bi)) { bv = v; bi = i; }
        }
        int m = mr + tid;
        if (m < M) idxOut[m] = bi;
    }
}

// ---------------- update: gather + linear-interp, f_hat += h, f_rest -= h, loss partials
__global__ void k_update(const int* __restrict__ idx, const float* __restrict__ emb,
                         const float* __restrict__ f, float* __restrict__ f_hat,
                         float* __restrict__ f_rest, float* __restrict__ qpart, int pn) {
    __shared__ float red[256];
    int bn = blockIdx.x;           // 0..B*N-1
    int b = bn / NN, n = bn % NN;
    int c = threadIdx.x;

    float scale = (float)pn / (float)NN;
    float src = ((float)n + 0.5f) * scale - 0.5f;
    src = src > 0.0f ? src : 0.0f;
    int i0 = (int)src;
    int i1 = i0 + 1 < pn ? i0 + 1 : pn - 1;
    float w = src - (float)i0;

    int id0 = idx[b * pn + i0];
    int id1 = idx[b * pn + i1];
    float h = emb[(size_t)id0 * CC + c] * (1.0f - w) + emb[(size_t)id1 * CC + c] * w;

    size_t off = (size_t)bn * CC + c;
    float fh = f_hat[off] + h;
    f_hat[off] = fh;
    f_rest[off] -= h;
    float diff = fh - f[off];
    red[c] = diff * diff;
    __syncthreads();
    for (int s = 128; s > 0; s >>= 1) {
        if (c < s) red[c] += red[c + s];
        __syncthreads();
    }
    if (c == 0) qpart[bn] = red[0];
}

// ---------------- deterministic fixed-order reduction of loss partials ----------------
__global__ void k_reduce(const float* __restrict__ qpart, float* __restrict__ qacc) {
    __shared__ float red[256];
    int t = threadIdx.x;
    float s = 0.0f;
    for (int i = t; i < BB * NN; i += 256) s += qpart[i];
    red[t] = s;
    __syncthreads();
    for (int k = 128; k > 0; k >>= 1) {
        if (t < k) red[t] += red[t + k];
        __syncthreads();
    }
    if (t == 0) *qacc += red[0];
}

__global__ void k_final(const float* __restrict__ qacc, float* __restrict__ out) {
    float qlat = *qacc / ((float)NUMEL * (float)SN);
    out[NUMEL + 0] = 0.25f * qlat;   // commit
    out[NUMEL + 1] = qlat;           // qlat
}

extern "C" void kernel_launch(void* const* d_in, const int* in_sizes, int n_in,
                              void* d_out, int out_size, void* d_ws, size_t ws_size,
                              hipStream_t stream) {
    const float* f      = (const float*)d_in[0];  // [B,N,C]
    const float* base   = (const float*)d_in[1];  // [K,C]
    const float* proj_w = (const float*)d_in[2];  // [C,C]
    const float* proj_b = (const float*)d_in[3];  // [C]
    float* out = (float*)d_out;                   // [B*N*C] f_hat + commit + qlat

    // Workspace layout (bytes), ~25.3 MB total
    char* ws = (char*)d_ws;
    float*          emb     = (float*)(ws);                               // 4 MB
    unsigned short* embT_hi = (unsigned short*)(ws + (4u << 20));         // 2 MB
    unsigned short* embT_lo = (unsigned short*)(ws + (6u << 20));         // 2 MB
    float*          esq     = (float*)(ws + (8u << 20));                  // 16 KB
    float*          f_rest  = (float*)(ws + (8u << 20) + (64u << 10));    // 8 MB
    unsigned short* q_hi    = (unsigned short*)(ws + (16u << 20) + (64u << 10)); // 4 MB
    unsigned short* q_lo    = (unsigned short*)(ws + (20u << 20) + (64u << 10)); // 4 MB
    int*            idx     = (int*)(ws + (24u << 20) + (64u << 10));     // 32 KB
    float*          qacc    = (float*)(ws + (24u << 20) + (96u << 10));   // 4 B (padded)
    float*          qpart   = (float*)(ws + (24u << 20) + (128u << 10));  // 32 KB

    k_init<<<NUMEL / 256, 256, 0, stream>>>(out, f, f_rest, qacc);
    k_emb<<<KK, 256, 0, stream>>>(base, proj_w, proj_b, emb, embT_hi, embT_lo, esq);

    for (int s = 0; s < SN; ++s) {
        int pn = 1 << s;
        int M = BB * pn;
        int Mpad = (M + 15) & ~15;
        k_pool<<<Mpad, 256, 0, stream>>>(f_rest, q_hi, q_lo, pn, M);
        k_dist<<<Mpad / 16, 256, 0, stream>>>(q_hi, q_lo, embT_hi, embT_lo, esq, idx, M);
        k_update<<<BB * NN, 256, 0, stream>>>(idx, emb, f, out, f_rest, qpart, pn);
        k_reduce<<<1, 256, 0, stream>>>(qpart, qacc);
    }
    k_final<<<1, 1, 0, stream>>>(qacc, out);
}
